// GATLayer_56702158242138
// MI455X (gfx1250) — matrix operand order repaired and checked by reading, not compiled, not run on hardware
//
#include <hip/hip_runtime.h>

typedef __attribute__((ext_vector_type(16))) _Float16 v16h;
typedef __attribute__((ext_vector_type(8)))  float    v8f;

#define IN_DIM  128
#define OUT_DIM 128

// h = x @ W_lin^T via v_wmma_f32_16x16x32_f16.
// Each wave computes one 16x16 tile of h. 8 waves/block cover the 8 N-tiles
// (OUT_DIM=128) of one 16-row M-tile. Writes h to both d_ws (scatter source)
// and d_out (initial value of h_prime).
__global__ __launch_bounds__(256)
void gat_gemm_wmma(const float* __restrict__ x, const float* __restrict__ W,
                   float* __restrict__ h, float* __restrict__ out, int n_nodes)
{
    const int lane = threadIdx.x & 31;
    const int wave = threadIdx.x >> 5;      // 0..7 -> N tile
    const int m0   = blockIdx.x * 16;
    const int n0   = wave * 16;
    const int half = lane >> 4;             // 0: lanes 0-15, 1: lanes 16-31
    const int l15  = lane & 15;

    int arow = m0 + l15;
    if (arow >= n_nodes) arow = n_nodes - 1;   // clamp (50000 % 16 == 0 anyway)
    const float* xrow = x + (size_t)arow * IN_DIM;
    const float* wrow = W + (size_t)(n0 + l15) * IN_DIM;

    v8f acc = {0.f, 0.f, 0.f, 0.f, 0.f, 0.f, 0.f, 0.f};

    #pragma unroll
    for (int k = 0; k < IN_DIM; k += 32) {
        // A-matrix lane data: K = k + 8*half + [0..7]  and  k + 16 + 8*half + [0..7]
        const float4 a0 = *(const float4*)(xrow + k + 8 * half);
        const float4 a1 = *(const float4*)(xrow + k + 8 * half + 4);
        const float4 a2 = *(const float4*)(xrow + k + 16 + 8 * half);
        const float4 a3 = *(const float4*)(xrow + k + 16 + 8 * half + 4);
        // B-matrix lane data: K = k + 8*half + [0..7]  and  k + 16 + 8*half + [0..7], column n0+l15 == row of W
        const float4 b0 = *(const float4*)(wrow + k + 8 * half);
        const float4 b1 = *(const float4*)(wrow + k + 8 * half + 4);
        const float4 b2 = *(const float4*)(wrow + k + 16 + 8 * half);
        const float4 b3 = *(const float4*)(wrow + k + 16 + 8 * half + 4);

        const float af[16] = {a0.x, a0.y, a0.z, a0.w, a1.x, a1.y, a1.z, a1.w,
                              a2.x, a2.y, a2.z, a2.w, a3.x, a3.y, a3.z, a3.w};
        const float bf[16] = {b0.x, b0.y, b0.z, b0.w, b1.x, b1.y, b1.z, b1.w,
                              b2.x, b2.y, b2.z, b2.w, b3.x, b3.y, b3.z, b3.w};
        v16h A, B;
        #pragma unroll
        for (int i = 0; i < 16; ++i) {
            A[i] = (_Float16)af[i];
            B[i] = (_Float16)bf[i];
        }

        // (neg_a, A, neg_b, B, c_mod, C, reuse_a, reuse_b)
        acc = __builtin_amdgcn_wmma_f32_16x16x32_f16(
            false, A, false, B, (short)0, acc, false, false);
    }

    // D layout: acc[r] -> row m0 + 8*half + r, col n0 + l15
    #pragma unroll
    for (int r = 0; r < 8; ++r) {
        const int row = m0 + 8 * half + r;
        if (row < n_nodes) {
            const size_t idx = (size_t)row * OUT_DIM + n0 + l15;
            h[idx]   = acc[r];
            out[idx] = acc[r];
        }
    }
}

// alpha == softmax over a singleton axis == 1.0 exactly, so:
//   out[src[e]] += h[dst[e]]   for every edge e.
// One wave32 per edge: lanes cover the 128-float row as float4 reads and
// 4 global f32 atomic adds each. h (25.6 MB) and out (25.6 MB) are L2-resident
// (192 MB L2), so this is an L2 atomic-throughput workload.
__global__ __launch_bounds__(256)
void gat_scatter(const long long* __restrict__ edge, const float* __restrict__ h,
                 float* __restrict__ out, int n_edges)
{
    const int e = blockIdx.x * 8 + (int)(threadIdx.x >> 5);
    if (e >= n_edges) return;
    const int lane = threadIdx.x & 31;

    const long long s = edge[e];                        // edge_index[0, e]
    const long long d = edge[(size_t)n_edges + e];      // edge_index[1, e]

    const float4 v = *(const float4*)(h + (size_t)d * OUT_DIM + lane * 4);
    float* o = out + (size_t)s * OUT_DIM + lane * 4;

    unsafeAtomicAdd(o + 0, v.x);
    unsafeAtomicAdd(o + 1, v.y);
    unsafeAtomicAdd(o + 2, v.z);
    unsafeAtomicAdd(o + 3, v.w);
}

extern "C" void kernel_launch(void* const* d_in, const int* in_sizes, int n_in,
                              void* d_out, int out_size, void* d_ws, size_t ws_size,
                              hipStream_t stream) {
    const float*     x    = (const float*)d_in[0];       // (N_NODES, 128) f32
    const long long* edge = (const long long*)d_in[1];   // (2, N_EDGES) i64
    const float*     W    = (const float*)d_in[2];       // (128, 128) f32
    // d_in[3] = W_attn: mathematically dead (softmax over singleton axis == 1)

    float* out = (float*)d_out;
    float* h   = (float*)d_ws;                           // needs 50000*128*4 = 25.6 MB

    const int n_nodes = in_sizes[0] / IN_DIM;
    const int n_edges = in_sizes[1] / 2;

    const int gemm_blocks = (n_nodes + 15) / 16;
    gat_gemm_wmma<<<gemm_blocks, 256, 0, stream>>>(x, W, h, out, n_nodes);

    const int scat_blocks = (n_edges + 7) / 8;           // 8 waves (edges) per block
    gat_scatter<<<scat_blocks, 256, 0, stream>>>(edge, h, out, n_edges);
}
